// Model_34471407517969
// MI455X (gfx1250) — matrix operand order repaired and checked by reading, not compiled, not run on hardware
//
#include <hip/hip_runtime.h>
#include <hip/hip_bf16.h>
#include <math.h>

// ---------------- problem dims ----------------
#define H     400
#define Hp    416        // H padded to multiple of 32 for WMMA K
#define B     64
#define S     256
#define BERT  768
#define G4H   1600       // 4*H gate width
#define NK    12
#define NM    7
#define CPL   9
#define MPL   129
#define NPC   8
#define NC    32

#define NWG   64         // persistent grid: 64 workgroups
#define TPB   256        // 8 wave32 waves per WG

typedef __attribute__((ext_vector_type(16))) __bf16 v16bf;
typedef __attribute__((ext_vector_type(8)))  __bf16 v8bf;
typedef __attribute__((ext_vector_type(8)))  float  v8f;

// ---------------- small device helpers ----------------
__device__ __forceinline__ __bf16 f2bf(float f) {
    union { float f; unsigned u; } v; v.f = f;
    unsigned r = (v.u + 0x7fffu + ((v.u >> 16) & 1u)) >> 16;   // RNE
    union { unsigned short s; __bf16 b; } o; o.s = (unsigned short)r;
    return o.b;
}
__device__ __forceinline__ float sigm(float x) { return 1.f / (1.f + __expf(-x)); }

// A fragment: 16x32 bf16 tile, row-major source (batch rows), contiguous k order (k = 16*half + s).
// lanes 0-15: M=lane, K=0..15; lanes 16-31: M=lane-16, K=16..31
__device__ __forceinline__ v16bf load_frag_A(const __bf16* A, int lda, int lane) {
    int half = lane >> 4, m = lane & 15;
    const __bf16* p = A + m * lda + half * 16;
    v8bf lo = *(const v8bf*)p;
    v8bf hi = *(const v8bf*)(p + 8);
    v16bf r;
#pragma unroll
    for (int i = 0; i < 8; ++i) { r[i] = lo[i]; r[8 + i] = hi[i]; }
    return r;
}
// B fragment: 32x16 bf16 (K x N). Weight is stored row-major [N x K] (torch
// layout), which is exactly B^T, so each lane does ONE contiguous 32B load:
// lanes 0-15: N=lane, K=0..15; lanes 16-31: N=lane-16, K=16..31
__device__ __forceinline__ v16bf load_frag_B(const __bf16* W, int ldw, int lane) {
    int half = lane >> 4, n = lane & 15;
    return *(const v16bf*)(W + n * ldw + half * 16);
}

// One WMMA GEMM "stage": D[64 x 16*Ntiles] = A1*W1^T (+ A2*W2^T) + bias, over
// all waves in the persistent grid. K1,K2 multiples of 32. C/D layout per ISA:
// lane half/n, VGPR r -> row m0+half*8+r, col n0+n.
__device__ void wmma_stage(const __bf16* A1, int lda1, int K1,
                           const __bf16* W1, int ldw1,
                           const __bf16* A2, int lda2, int K2,
                           const __bf16* W2, int ldw2,
                           const float* bias, float* D, int ldd,
                           int Ntiles, int relu, int wid, int nwv)
{
    int lane = threadIdx.x & 31;
    int half = lane >> 4, nl = lane & 15;
    int ntiles = 4 * Ntiles;                 // M=64 -> 4 M-tiles
    for (int t = wid; t < ntiles; t += nwv) {
        int m0 = (t & 3) << 4;
        int n0 = (t >> 2) << 4;
        float bv = bias[n0 + nl];
        v8f c;
#pragma unroll
        for (int r = 0; r < 8; ++r) c[r] = bv;  // accumulator preloaded with bias
        const __bf16* a1 = A1 + m0 * lda1;
        const __bf16* w1 = W1 + n0 * ldw1;
        for (int k = 0; k < K1; k += 32) {
            __builtin_prefetch(w1 + k + 32, 0, 1);       // global_prefetch_b8
            v16bf a = load_frag_A(a1 + k, lda1, lane);
            v16bf b = load_frag_B(w1 + k, ldw1, lane);
            c = __builtin_amdgcn_wmma_f32_16x16x32_bf16(false, a, false, b,
                                                        (short)0, c, false, false);
        }
        if (A2) {
            const __bf16* a2 = A2 + m0 * lda2;
            const __bf16* w2 = W2 + n0 * ldw2;
            for (int k = 0; k < K2; k += 32) {
                v16bf a = load_frag_A(a2 + k, lda2, lane);
                v16bf b = load_frag_B(w2 + k, ldw2, lane);
                c = __builtin_amdgcn_wmma_f32_16x16x32_bf16(false, a, false, b,
                                                            (short)0, c, false, false);
            }
        }
#pragma unroll
        for (int r = 0; r < 8; ++r) {
            float v = c[r];
            if (relu) v = fmaxf(v, 0.f);
            D[(m0 + half * 8 + r) * ldd + n0 + nl] = v;
        }
    }
}

// Grid-wide sense-reversal barrier (persistent kernel, all NWG blocks resident)
__device__ __forceinline__ void gbar(int* cnt, int* phase) {
    __syncthreads();
    if (threadIdx.x == 0) {
        __threadfence();
        int ph = __hip_atomic_load(phase, __ATOMIC_RELAXED, __HIP_MEMORY_SCOPE_AGENT);
        int arrived = __hip_atomic_fetch_add(cnt, 1, __ATOMIC_ACQ_REL, __HIP_MEMORY_SCOPE_AGENT);
        if (arrived == NWG - 1) {
            __hip_atomic_store(cnt, 0, __ATOMIC_RELAXED, __HIP_MEMORY_SCOPE_AGENT);
            __hip_atomic_fetch_add(phase, 1, __ATOMIC_RELEASE, __HIP_MEMORY_SCOPE_AGENT);
        } else {
            while (__hip_atomic_load(phase, __ATOMIC_ACQUIRE, __HIP_MEMORY_SCOPE_AGENT) == ph)
                __builtin_amdgcn_s_sleep(2);
        }
    }
    __syncthreads();
}

// LSTM elementwise: gates [B x 4H] (i,f,g,o) -> update h,c, emit bf16 h (padded)
__device__ void lstm_elem_stage(const float* gates, float* h, float* c,
                                __bf16* hbf, int tid, int nth)
{
    for (int i = tid; i < B * H; i += nth) {
        int b = i / H, j = i % H;
        const float* g = gates + b * G4H;
        float ig = sigm(g[j]);
        float fg = sigm(g[H + j]);
        float gg = tanhf(g[2 * H + j]);
        float og = sigm(g[3 * H + j]);
        float ccv = fg * c[i] + ig * gg;
        float hh  = og * tanhf(ccv);
        c[i] = ccv; h[i] = hh;
        hbf[b * Hp + j] = f2bf(hh);
    }
}

// ---------------- prep kernels ----------------
__global__ void k_f2bf_mat(const float* src, __bf16* dst, int N, int K, int Kp) {
    long i = (long)blockIdx.x * blockDim.x + threadIdx.x;
    if (i >= (long)N * Kp) return;
    int n = (int)(i / Kp), k = (int)(i % Kp);
    dst[i] = (k < K) ? f2bf(src[(long)n * K + k]) : f2bf(0.f);
}
__global__ void k_addvec(const float* a, const float* b, float* o, int n) {
    int i = blockIdx.x * blockDim.x + threadIdx.x;
    if (i < n) o[i] = a[i] + b[i];
}
__global__ void k_copy2d(const float* X, int ldx, float* Y, int ldy, int M, int N) {
    int i = blockIdx.x * blockDim.x + threadIdx.x;
    if (i >= M * N) return;
    int m = i / N, n = i % N;
    Y[m * ldy + n] = X[m * ldx + n];
}
__global__ void k_state_init(const float* hxC, const float* cxC,
                             const float* hxM, const float* cxM,
                             float* hF, float* cF, float* hB, float* cB,
                             __bf16* hFbf, __bf16* hBbf,
                             float* hc, float* cc, float* hm, float* cm,
                             float* me, __bf16* minbf, int* cnt, int* phase)
{
    int i = blockIdx.x * blockDim.x + threadIdx.x;
    int nth = gridDim.x * blockDim.x;
    for (int j = i; j < B * H; j += nth) {
        hF[j] = 0.f; cF[j] = 0.f; hB[j] = 0.f; cB[j] = 0.f;
        hc[j] = hxC[j]; cc[j] = cxC[j]; hm[j] = hxM[j]; cm[j] = cxM[j];
        me[j] = 0.f;
    }
    for (int j = i; j < B * Hp; j += nth) {
        hFbf[j] = f2bf(0.f); hBbf[j] = f2bf(0.f); minbf[j] = f2bf(0.f);
    }
    if (i == 0) { *cnt = 0; *phase = 0; }
}

// ---------------- encoder: persistent bidirectional LSTM scan ----------------
__global__ void k_encoder(const __bf16* xbf,
                          const __bf16* WiF, const __bf16* WhF, const float* biasF,
                          const __bf16* WiB, const __bf16* WhB, const float* biasB,
                          float* hF, float* cF, __bf16* hFbf,
                          float* hB, float* cB, __bf16* hBbf,
                          float* gatesF, float* gatesB, float* temp,
                          int* cnt, int* phase)
{
    int tid = blockIdx.x * blockDim.x + threadIdx.x;
    int nth = gridDim.x * blockDim.x;
    int wid = tid >> 5, nwv = nth >> 5;
    for (int s = 0; s < S; ++s) {
        // gates = x_s*Wi^T + h*Wh^T + (bi+bh); x layout [B,S,BERT] -> lda=S*BERT
        wmma_stage(xbf + (long)s * BERT, S * BERT, BERT, WiF, BERT,
                   hFbf, Hp, Hp, WhF, Hp, biasF, gatesF, G4H, G4H / 16, 0, wid, nwv);
        wmma_stage(xbf + (long)(S - 1 - s) * BERT, S * BERT, BERT, WiB, BERT,
                   hBbf, Hp, Hp, WhB, Hp, biasB, gatesB, G4H, G4H / 16, 0, wid, nwv);
        gbar(cnt, phase);
        lstm_elem_stage(gatesF, hF, cF, hFbf, tid, nth);
        lstm_elem_stage(gatesB, hB, cB, hBbf, tid, nth);
        gbar(cnt, phase);
    }
    for (int i = tid; i < B * H; i += nth) temp[i] = hF[i] + hB[i];
}

// ---------------- small dense heads (tiny GEMMs, VALU is fine) ----------------
__global__ void k_gemm_dot(const float* X, int ldx, const float* W, const float* bias,
                           float* Y, int ldy, int M, int N, int K, int act)
{
    int i = blockIdx.x * blockDim.x + threadIdx.x;
    if (i >= M * N) return;
    int m = i / N, n = i % N;
    const float* x = X + (long)m * ldx;
    const float* w = W + (long)n * K;
    float acc = bias ? bias[n] : 0.f;
    for (int k = 0; k < K; ++k) acc += x[k] * w[k];
    if (act) acc = fmaxf(acc, 0.f);
    Y[(long)m * ldy + n] = acc;
}
__global__ void k_kl(const float* mu, const float* lv, float* out) {
    __shared__ float red[256];
    float s = 0.f;
    for (int i = threadIdx.x; i < B * H; i += 256) {
        float m = mu[i], l = lv[i];
        s += m * m + __expf(l) - l - 1.f;
    }
    red[threadIdx.x] = s; __syncthreads();
    for (int o = 128; o > 0; o >>= 1) {
        if (threadIdx.x < o) red[threadIdx.x] += red[threadIdx.x + o];
        __syncthreads();
    }
    if (threadIdx.x == 0) out[0] = 0.5f * red[0] / (float)(B * H);
}

// ---------------- decoder: persistent chord/note autoregressive scan ----------
__global__ void k_decoder(const __bf16* cWi, const __bf16* cWh, const float* biasC,
                          const __bf16* mWi, const __bf16* mWh, const float* biasM,
                          const __bf16* chW1, const float* chb1,
                          const float* chW2, const float* chb2,
                          const __bf16* meW1, const float* meb1,
                          const float* meW2, const float* meb2,
                          const float* chord_table, const float* melody_table,
                          float* hc, float* cc, __bf16* hcbf,
                          float* hm, float* cm, __bf16* hmbf,
                          __bf16* cinbf, __bf16* minbf,
                          float* ce, float* me, float* hid, float* gates,
                          int* idxbuf, float* out_ch, float* out_nt,
                          int* cnt, int* phase)
{
    int tid = blockIdx.x * blockDim.x + threadIdx.x;
    int nth = gridDim.x * blockDim.x;
    int wid = tid >> 5, nwv = nth >> 5;
    for (int t = 0; t < NC; ++t) {
        // chord LSTM cell
        wmma_stage(cinbf, Hp, Hp, cWi, Hp, hcbf, Hp, Hp, cWh, Hp,
                   biasC, gates, G4H, G4H / 16, 0, wid, nwv);
        gbar(cnt, phase);
        lstm_elem_stage(gates, hc, cc, hcbf, tid, nth);
        gbar(cnt, phase);
        // chord MLP hidden (relu) via WMMA
        wmma_stage(hcbf, Hp, Hp, chW1, Hp, nullptr, 0, 0, nullptr, 0,
                   chb1, hid, H, H / 16, 1, wid, nwv);
        gbar(cnt, phase);
        // cp = hid @ chW2^T + b2  -> out
        for (int i = tid; i < B * CPL; i += nth) {
            int b = i / CPL, n = i % CPL;
            const float* x = hid + b * H; const float* w = chW2 + n * H;
            float a = chb2[n];
            for (int k = 0; k < H; ++k) a += x[k] * w[k];
            out_ch[b * NC * CPL + t * CPL + n] = a;
        }
        gbar(cnt, phase);
        for (int b = tid; b < B; b += nth) {
            const float* cp = out_ch + b * NC * CPL + t * CPL;
            int best = 0; float bv = cp[0];
            for (int j = 1; j < CPL; ++j) if (cp[j] > bv) { bv = cp[j]; best = j; }
            idxbuf[b] = best;
        }
        gbar(cnt, phase);
        // ce = chord_table[argmax]; cin for next chord = ce; minbf = bf16(me+ce)
        for (int i = tid; i < B * Hp; i += nth) {
            int b = i / Hp, j = i % Hp;
            float v = (j < H) ? chord_table[idxbuf[b] * H + j] : 0.f;
            if (j < H) ce[b * H + j] = v;
            cinbf[i] = f2bf(v);
            float mi = (j < H) ? (me[b * H + j] + v) : 0.f;
            minbf[i] = f2bf(mi);
        }
        gbar(cnt, phase);
        for (int n = 0; n < NPC; ++n) {
            wmma_stage(minbf, Hp, Hp, mWi, Hp, hmbf, Hp, Hp, mWh, Hp,
                       biasM, gates, G4H, G4H / 16, 0, wid, nwv);
            gbar(cnt, phase);
            lstm_elem_stage(gates, hm, cm, hmbf, tid, nth);
            gbar(cnt, phase);
            wmma_stage(hmbf, Hp, Hp, meW1, Hp, nullptr, 0, 0, nullptr, 0,
                       meb1, hid, H, H / 16, 1, wid, nwv);
            gbar(cnt, phase);
            float* mpbase = out_nt + (t * NPC + n) * MPL;   // [B][NC*NPC][MPL]
            for (int i = tid; i < B * MPL; i += nth) {
                int b = i / MPL, m = i % MPL;
                const float* x = hid + b * H; const float* w = meW2 + m * H;
                float a = meb2[m];
                for (int k = 0; k < H; ++k) a += x[k] * w[k];
                mpbase[(long)b * (NC * NPC * MPL) + m] = a;
            }
            gbar(cnt, phase);
            for (int b = tid; b < B; b += nth) {
                const float* mp = mpbase + (long)b * (NC * NPC * MPL);
                int best = 0; float bv = mp[0];
                for (int j = 1; j < MPL; ++j) if (mp[j] > bv) { bv = mp[j]; best = j; }
                idxbuf[b] = best;
            }
            gbar(cnt, phase);
            for (int i = tid; i < B * Hp; i += nth) {
                int b = i / Hp, j = i % Hp;
                float mv = (j < H) ? melody_table[idxbuf[b] * H + j] : 0.f;
                if (j < H) me[b * H + j] = mv;
                float mi = (j < H) ? (mv + ce[b * H + j]) : 0.f;
                minbf[i] = f2bf(mi);
            }
            gbar(cnt, phase);
        }
    }
}

// ---------------- host launcher ----------------
#define GRID(n) dim3((unsigned)(((long)(n) + 255) / 256))

extern "C" void kernel_launch(void* const* d_in, const int* in_sizes, int n_in,
                              void* d_out, int out_size, void* d_ws, size_t ws_size,
                              hipStream_t stream)
{
    (void)in_sizes; (void)n_in; (void)out_size; (void)ws_size;
    const float* IN[66];
    for (int i = 0; i < 66; ++i) IN[i] = (const float*)d_in[i];
    // input order: 0 input, 1-4 hx/cx, 5-12 enc_f/enc_b (Wi,Wh,bi,bh),
    // 13-16 mean/var, 17-36 five MLPs (W1,b1,W2,b2), 37-44 embeddings,
    // 45-46 down, 47-54 clstm/mlstm, 55 chord_table, 56 melody_table,
    // 57-60 chord_mlp, 61-64 mel_mlp, 65 num_chords
    float* out = (float*)d_out;
    const size_t OFS_NT  = (size_t)B * NC * CPL;             // 18432
    const size_t OFS_BPM = OFS_NT + (size_t)B * NC * NPC * MPL;
    const size_t OFS_KEY = OFS_BPM + B;
    const size_t OFS_MOD = OFS_KEY + (size_t)B * NK;
    const size_t OFS_VAL = OFS_MOD + (size_t)B * NM;
    const size_t OFS_EN  = OFS_VAL + B;
    const size_t OFS_KL  = OFS_EN + B;

    char* wsb = (char*)d_ws;
    size_t off = 0;
    auto alloc = [&](size_t bytes) -> void* {
        void* p = wsb + off; off = (off + bytes + 255) & ~(size_t)255; return p;
    };
    // bf16 buffers
    __bf16* xbf   = (__bf16*)alloc((size_t)B * S * BERT * 2);
    __bf16* WiF   = (__bf16*)alloc((size_t)G4H * BERT * 2);
    __bf16* WiB   = (__bf16*)alloc((size_t)G4H * BERT * 2);
    __bf16* WhF   = (__bf16*)alloc((size_t)G4H * Hp * 2);
    __bf16* WhB   = (__bf16*)alloc((size_t)G4H * Hp * 2);
    __bf16* cWi   = (__bf16*)alloc((size_t)G4H * Hp * 2);
    __bf16* cWh   = (__bf16*)alloc((size_t)G4H * Hp * 2);
    __bf16* mWi   = (__bf16*)alloc((size_t)G4H * Hp * 2);
    __bf16* mWh   = (__bf16*)alloc((size_t)G4H * Hp * 2);
    __bf16* chW1  = (__bf16*)alloc((size_t)H * Hp * 2);
    __bf16* meW1  = (__bf16*)alloc((size_t)H * Hp * 2);
    __bf16* hFbf  = (__bf16*)alloc((size_t)B * Hp * 2);
    __bf16* hBbf  = (__bf16*)alloc((size_t)B * Hp * 2);
    __bf16* hcbf  = (__bf16*)alloc((size_t)B * Hp * 2);
    __bf16* hmbf  = (__bf16*)alloc((size_t)B * Hp * 2);
    __bf16* cinbf = (__bf16*)alloc((size_t)B * Hp * 2);
    __bf16* minbf = (__bf16*)alloc((size_t)B * Hp * 2);
    // f32 buffers
    float* biasF = (float*)alloc(G4H * 4);
    float* biasB = (float*)alloc(G4H * 4);
    float* biasC = (float*)alloc(G4H * 4);
    float* biasM = (float*)alloc(G4H * 4);
    float* hF = (float*)alloc(B * H * 4);  float* cF = (float*)alloc(B * H * 4);
    float* hB = (float*)alloc(B * H * 4);  float* cB = (float*)alloc(B * H * 4);
    float* hc = (float*)alloc(B * H * 4);  float* cc = (float*)alloc(B * H * 4);
    float* hm = (float*)alloc(B * H * 4);  float* cm = (float*)alloc(B * H * 4);
    float* mu = (float*)alloc(B * H * 4);  float* lv = (float*)alloc(B * H * 4);
    float* temp = (float*)alloc(B * H * 4);
    float* z2   = (float*)alloc(B * H * 4);
    float* hid  = (float*)alloc(B * H * 4);
    float* ce   = (float*)alloc(B * H * 4);
    float* me   = (float*)alloc(B * H * 4);
    float* gatesF = (float*)alloc((size_t)B * G4H * 4);
    float* gatesB = (float*)alloc((size_t)B * G4H * 4);
    float* cat  = (float*)alloc((size_t)B * 5 * H * 4);
    int* idxbuf = (int*)alloc(B * 4);
    int* barcnt = (int*)alloc(4);
    int* barph  = (int*)alloc(4);

    // -------- prep: bf16 conversions, biases, state init --------
    k_f2bf_mat<<<GRID((long)B * S * BERT), 256, 0, stream>>>(IN[0], xbf, B * S, BERT, BERT);
    k_f2bf_mat<<<GRID((long)G4H * BERT), 256, 0, stream>>>(IN[5], WiF, G4H, BERT, BERT);
    k_f2bf_mat<<<GRID((long)G4H * Hp),  256, 0, stream>>>(IN[6], WhF, G4H, H, Hp);
    k_f2bf_mat<<<GRID((long)G4H * BERT), 256, 0, stream>>>(IN[9], WiB, G4H, BERT, BERT);
    k_f2bf_mat<<<GRID((long)G4H * Hp),  256, 0, stream>>>(IN[10], WhB, G4H, H, Hp);
    k_f2bf_mat<<<GRID((long)G4H * Hp),  256, 0, stream>>>(IN[47], cWi, G4H, H, Hp);
    k_f2bf_mat<<<GRID((long)G4H * Hp),  256, 0, stream>>>(IN[48], cWh, G4H, H, Hp);
    k_f2bf_mat<<<GRID((long)G4H * Hp),  256, 0, stream>>>(IN[51], mWi, G4H, H, Hp);
    k_f2bf_mat<<<GRID((long)G4H * Hp),  256, 0, stream>>>(IN[52], mWh, G4H, H, Hp);
    k_f2bf_mat<<<GRID((long)H * Hp),    256, 0, stream>>>(IN[57], chW1, H, H, Hp);
    k_f2bf_mat<<<GRID((long)H * Hp),    256, 0, stream>>>(IN[61], meW1, H, H, Hp);
    k_addvec<<<GRID(G4H), 256, 0, stream>>>(IN[7], IN[8], biasF, G4H);
    k_addvec<<<GRID(G4H), 256, 0, stream>>>(IN[11], IN[12], biasB, G4H);
    k_addvec<<<GRID(G4H), 256, 0, stream>>>(IN[49], IN[50], biasC, G4H);
    k_addvec<<<GRID(G4H), 256, 0, stream>>>(IN[53], IN[54], biasM, G4H);
    k_state_init<<<GRID(B * Hp), 256, 0, stream>>>(IN[1], IN[2], IN[3], IN[4],
        hF, cF, hB, cB, hFbf, hBbf, hc, cc, hm, cm, me, minbf, barcnt, barph);
    k_f2bf_mat<<<GRID(B * Hp), 256, 0, stream>>>(IN[1], hcbf, B, H, Hp);
    k_f2bf_mat<<<GRID(B * Hp), 256, 0, stream>>>(IN[3], hmbf, B, H, Hp);

    // -------- encoder (persistent, WMMA) --------
    k_encoder<<<NWG, TPB, 0, stream>>>(xbf, WiF, WhF, biasF, WiB, WhB, biasB,
                                       hF, cF, hFbf, hB, cB, hBbf,
                                       gatesF, gatesB, temp, barcnt, barph);

    // -------- VAE head (tiny GEMMs) --------
    k_gemm_dot<<<GRID(B * H), 256, 0, stream>>>(temp, H, IN[13], IN[14], mu, H, B, H, H, 0);
    k_gemm_dot<<<GRID(B * H), 256, 0, stream>>>(temp, H, IN[15], IN[16], lv, H, B, H, H, 0);
    k_kl<<<1, 256, 0, stream>>>(mu, lv, out + OFS_KL);
    // key/mode/bpm/val/en MLPs (shared hidden scratch, stream-ordered)
    k_gemm_dot<<<GRID(B * H), 256, 0, stream>>>(mu, H, IN[17], IN[18], hid, H, B, H, H, 1);
    k_gemm_dot<<<GRID(B * NK), 256, 0, stream>>>(hid, H, IN[19], IN[20], out + OFS_KEY, NK, B, NK, H, 0);
    k_gemm_dot<<<GRID(B * H), 256, 0, stream>>>(mu, H, IN[21], IN[22], hid, H, B, H, H, 1);
    k_gemm_dot<<<GRID(B * NM), 256, 0, stream>>>(hid, H, IN[23], IN[24], out + OFS_MOD, NM, B, NM, H, 0);
    k_gemm_dot<<<GRID(B * H), 256, 0, stream>>>(mu, H, IN[25], IN[26], hid, H, B, H, H, 1);
    k_gemm_dot<<<GRID(B), 256, 0, stream>>>(hid, H, IN[27], IN[28], out + OFS_BPM, 1, B, 1, H, 0);
    k_gemm_dot<<<GRID(B * H), 256, 0, stream>>>(mu, H, IN[29], IN[30], hid, H, B, H, H, 1);
    k_gemm_dot<<<GRID(B), 256, 0, stream>>>(hid, H, IN[31], IN[32], out + OFS_VAL, 1, B, 1, H, 0);
    k_gemm_dot<<<GRID(B * H), 256, 0, stream>>>(mu, H, IN[33], IN[34], hid, H, B, H, H, 1);
    k_gemm_dot<<<GRID(B), 256, 0, stream>>>(hid, H, IN[35], IN[36], out + OFS_EN, 1, B, 1, H, 0);
    // cat = [z, bpm_emb, mode_emb, val_emb, en_emb]  (B x 2000)
    k_copy2d<<<GRID(B * H), 256, 0, stream>>>(mu, H, cat, 5 * H, B, H);
    k_gemm_dot<<<GRID(B * H), 256, 0, stream>>>(out + OFS_BPM, 1, IN[37], IN[38], cat + H, 5 * H, B, H, 1, 0);
    k_gemm_dot<<<GRID(B * H), 256, 0, stream>>>(out + OFS_MOD, NM, IN[39], IN[40], cat + 2 * H, 5 * H, B, H, NM, 0);
    k_gemm_dot<<<GRID(B * H), 256, 0, stream>>>(out + OFS_VAL, 1, IN[41], IN[42], cat + 3 * H, 5 * H, B, H, 1, 0);
    k_gemm_dot<<<GRID(B * H), 256, 0, stream>>>(out + OFS_EN, 1, IN[43], IN[44], cat + 4 * H, 5 * H, B, H, 1, 0);
    k_gemm_dot<<<GRID(B * H), 256, 0, stream>>>(cat, 5 * H, IN[45], IN[46], z2, H, B, H, 5 * H, 0);
    k_f2bf_mat<<<GRID(B * Hp), 256, 0, stream>>>(z2, cinbf, B, H, Hp);

    // -------- decoder (persistent, WMMA) --------
    k_decoder<<<NWG, TPB, 0, stream>>>(cWi, cWh, biasC, mWi, mWh, biasM,
                                       chW1, IN[58], IN[59], IN[60],
                                       meW1, IN[62], IN[63], IN[64],
                                       IN[55], IN[56],
                                       hc, cc, hcbf, hm, cm, hmbf,
                                       cinbf, minbf, ce, me, hid, gatesF,
                                       idxbuf, out, out + OFS_NT, barcnt, barph);
}
